// FuzzyAutoma_non_mutex_8186207666312
// MI455X (gfx1250) — matrix operand order repaired and checked by reading, not compiled, not run on hardware
//
#include <hip/hip_runtime.h>
#include <stdint.h>

typedef _Float16 f16;
typedef __attribute__((ext_vector_type(16))) _Float16      v16h;
typedef __attribute__((ext_vector_type(8)))  float         v8f;
typedef __attribute__((ext_vector_type(4)))  unsigned int  v4u;
typedef __attribute__((ext_vector_type(8)))  int           v8i;
typedef __attribute__((ext_vector_type(4)))  int           v4i;

#define NUM_STATES  65536
#define NUM_SYMBOLS 64

// ---------------------------------------------------------------------------
// Pack each edge into 8 bytes: [63:32] = src (16 bits used), [31:0] = dst*64+sym
// 32 MB total -> stays L2-resident for all 128 timesteps.
// ---------------------------------------------------------------------------
__global__ __launch_bounds__(1024) void pack_edges(const int* __restrict__ src,
                                                   const int* __restrict__ sym,
                                                   const int* __restrict__ dst,
                                                   uint64_t* __restrict__ packed, int E) {
    int i = blockIdx.x * blockDim.x + threadIdx.x;
    if (i < E) {
        uint32_t gidx = ((uint32_t)dst[i] << 6) | (uint32_t)sym[i];
        packed[i] = ((uint64_t)(uint32_t)src[i] << 32) | (uint64_t)gidx;
    }
}

__global__ __launch_bounds__(1024) void zero_f32(float* __restrict__ p, int n) {
    int i = blockIdx.x * blockDim.x + threadIdx.x;
    if (i < n) p[i] = 0.0f;
}

__global__ __launch_bounds__(256) void init_state(float* __restrict__ s) {
    int i = blockIdx.x * blockDim.x + threadIdx.x;
    s[i] = (i == 0) ? 1.0f : 0.0f;
}

// ---------------------------------------------------------------------------
// Phase 1 (per step): G[dst*64+sym] += state[src] over all edges.
// Full 256 KB state vector is replicated into this WGP's LDS with one TDM
// tensor_load_to_lds (2D tile 256x256 f32), so every gather is a ds_load_b32.
// Scatter-adds are relaxed agent-scope f32 atomics (native global_atomic_add_f32,
// no-return) into the 16 MB row-major G buffer in L2 — 64x contention spreading
// vs. scattering into next[] directly.
// ---------------------------------------------------------------------------
__global__ __launch_bounds__(1024) void edge_accum(const uint64_t* __restrict__ edges,
                                                   const float* __restrict__ state,
                                                   float* __restrict__ G, int E) {
    __shared__ float s_state[NUM_STATES];   // 256 KB (<= 320 KB LDS per WGP)

    if (threadIdx.x < 32) {   // wave 0 issues the TDM descriptor (EXEC ignored by TDM)
        uint64_t ga  = (uint64_t)(uintptr_t)state;                 // 57-bit global addr
        uint32_t lds = (uint32_t)(uintptr_t)(&s_state[0]);         // LDS byte offset

        // D# group 0: count=1, lds_addr, global_addr, type=2 (bits 127:126)
        v4u g0;
        g0.x = 1u;                                   // count=1, gather off, user mode
        g0.y = lds;                                  // lds_addr [63:32]
        g0.z = (uint32_t)ga;                         // global_addr[31:0]
        g0.w = (uint32_t)(ga >> 32) | 0x80000000u;   // global_addr[56:32] | type=2<<30

        // D# group 1: data_size=4B, tensor 256x256, tile 256x256, stride0=256
        v8i g1;
        g1[0] = 0x20000;        // data_size=2 (4 bytes), mask=0, no pad/iterate
        g1[1] = 256 << 16;      // tensor_dim0[15:0]=256 at bits[63:48]
        g1[2] = 256 << 16;      // tensor_dim1[15:0]=256 at bits[95:80]
        g1[3] = 256 << 16;      // tile_dim0=256 at bits[127:112]
        g1[4] = 256;            // tile_dim1=256, tile_dim2=0
        g1[5] = 256;            // tensor_dim0_stride[31:0]=256
        g1[6] = 0;
        g1[7] = 0;

        v4i z4 = {};            // groups 2/3 unused for a 2D tensor
        v8i z8 = {};            // 5th arg on the 6-arg (clang-23) builtin
        __builtin_amdgcn_tensor_load_to_lds(g0, g1, z4, z4, z8, 0);
        __builtin_amdgcn_s_wait_tensorcnt(0);
    }
    __syncthreads();

    int idx    = blockIdx.x * blockDim.x + threadIdx.x;
    int stride = gridDim.x * blockDim.x;
    for (; idx < E; idx += stride) {
        uint64_t w = edges[idx];
        float s = s_state[(uint32_t)(w >> 32)];           // ds_load_b32 gather
        // Relaxed, agent-scope f32 fetch-add -> native global_atomic_add_f32
        // (no CAS loop, no pre-op return since the result is unused).
        __hip_atomic_fetch_add(&G[(uint32_t)w & 0x3FFFFFFu], s,
                               __ATOMIC_RELAXED, __HIP_MEMORY_SCOPE_AGENT);
    }
}

// ---------------------------------------------------------------------------
// Phase 2 (per step): next[d] = sum_k G[d][k] * action[k], done as WMMA:
// D(16x16) = A(16x32 f16: G tile) x B(32x16 f16: action bcast) + C, twice (K=64).
// One wave per 16-row dst tile; the wave also re-zeroes its G tile so the
// next timestep's phase 1 starts from a clean accumulator (no memset pass).
// VGPR layouts follow CDNA5 ISA (16-bit A 16x32; B rows striped;
// D: VGPR r / lanes<16 -> M=r, lanes>=16 -> M=r+8).
// ---------------------------------------------------------------------------
__device__ inline v16h pack16(float4 p0, float4 p1, float4 p2, float4 p3) {
    v16h a;
    a[0]=(f16)p0.x;  a[1]=(f16)p0.y;  a[2]=(f16)p0.z;  a[3]=(f16)p0.w;
    a[4]=(f16)p1.x;  a[5]=(f16)p1.y;  a[6]=(f16)p1.z;  a[7]=(f16)p1.w;
    a[8]=(f16)p2.x;  a[9]=(f16)p2.y;  a[10]=(f16)p2.z; a[11]=(f16)p2.w;
    a[12]=(f16)p3.x; a[13]=(f16)p3.y; a[14]=(f16)p3.z; a[15]=(f16)p3.w;
    return a;
}

__global__ __launch_bounds__(256) void reduce_wmma(float* __restrict__ G,
                                                   const float* __restrict__ act,
                                                   float* __restrict__ next) {
    int lane = threadIdx.x & 31;
    int wave = threadIdx.x >> 5;
    int tile = blockIdx.x * 8 + wave;       // 4096 tiles of 16 dst rows
    int d0   = tile << 4;
    int m    = lane & 15;
    int hi   = lane >> 4;                   // 0: K-lo half, 1: K-hi half
    int row  = d0 + m;
    int q    = hi << 1;                     // float4 index: 0 or 2 (k offset 0 or 8)

    const float4* gp = (const float4*)(G + ((size_t)row << 6));
    // A tile elements per documented 16-bit A 16x32 layout
    float4 x0 = gp[q + 0],  x1 = gp[q + 1];    // k  = kb .. kb+7
    float4 x2 = gp[q + 4],  x3 = gp[q + 5];    // k  = 16+kb ..
    float4 y0 = gp[q + 8],  y1 = gp[q + 9];    // k  = 32+kb ..
    float4 y2 = gp[q + 12], y3 = gp[q + 13];   // k  = 48+kb ..
    v16h a0 = pack16(x0, x1, x2, x3);
    v16h a1 = pack16(y0, y1, y2, y3);

    // B(32x16): B[k][n] = act[k] for all n. Lanes 0-15 hold K=0..7,16..23, lanes
    // 16-31 hold K=8..15,24..31 (two K per VGPR).
    int bb = hi ? 8 : 0;
    v16h b0, b1;
#pragma unroll
    for (int i = 0; i < 16; ++i) {
        b0[i] = (f16)act[bb + i + (i & 8)];
        b1[i] = (f16)act[bb + 32 + i + (i & 8)];
    }

    v8f c = {};
    c = __builtin_amdgcn_wmma_f32_16x16x32_f16(false, a0, false, b0, (short)0, c, false, false);
    c = __builtin_amdgcn_wmma_f32_16x16x32_f16(false, a1, false, b1, (short)0, c, false, false);

    // Every column of D equals next[]; lane 0 writes M=0..7, lane 16 writes M=8..15.
    if (m == 0) {
        int base = d0 + (hi << 3);
#pragma unroll
        for (int r = 0; r < 8; ++r) next[base + r] = c[r];
    }

    // Re-zero this tile's 16x64 G block for the next timestep.
    float* gz = G + ((size_t)d0 << 6);
#pragma unroll
    for (int i = 0; i < 32; ++i) gz[lane + 32 * i] = 0.0f;
}

// ---------------------------------------------------------------------------
// Workspace layout: [edges: E*8B = 32MB][G: 16MB][state0: 256KB][state1: 256KB]
// ---------------------------------------------------------------------------
extern "C" void kernel_launch(void* const* d_in, const int* in_sizes, int n_in,
                              void* d_out, int out_size, void* d_ws, size_t ws_size,
                              hipStream_t stream) {
    const float* prob = (const float*)d_in[0];
    const int*   src  = (const int*)d_in[1];
    const int*   sym  = (const int*)d_in[2];
    const int*   dst  = (const int*)d_in[3];
    const int E = in_sizes[1];
    const int T = in_sizes[0] / NUM_SYMBOLS;

    char*     ws    = (char*)d_ws;
    uint64_t* edges = (uint64_t*)ws;
    float*    G     = (float*)(ws + (size_t)E * 8);
    float*    s0    = (float*)((char*)G + (size_t)NUM_STATES * NUM_SYMBOLS * 4);
    float*    s1    = s0 + NUM_STATES;

    pack_edges<<<(E + 1023) / 1024, 1024, 0, stream>>>(src, sym, dst, edges, E);
    const int gn = NUM_STATES * NUM_SYMBOLS;
    zero_f32<<<(gn + 1023) / 1024, 1024, 0, stream>>>(G, gn);
    init_state<<<NUM_STATES / 256, 256, 0, stream>>>(s0);

    float* cur   = s0;
    float* other = s1;
    for (int t = 0; t < T; ++t) {
        edge_accum<<<256, 1024, 0, stream>>>(edges, cur, G, E);
        float* nxt = (t == T - 1) ? (float*)d_out : other;
        reduce_wmma<<<NUM_STATES / 16 / 8, 256, 0, stream>>>(
            G, prob + (size_t)t * NUM_SYMBOLS, nxt);
        other = cur;
        cur   = nxt;
    }
}
